// Encoder_10239202034038
// MI455X (gfx1250) — matrix operand order repaired and checked by reading, not compiled, not run on hardware
//
#include <hip/hip_runtime.h>

// ---------------- problem constants ----------------
static constexpr int kH   = 8;
static constexpr int kIN  = 512;
static constexpr int kSEQ = 1024;
static constexpr int kMID = 2048;
static constexpr int kOUT = 512;

// ---------------- WMMA / vector types ----------------
typedef __attribute__((ext_vector_type(16))) __bf16 v16bf;
typedef __attribute__((ext_vector_type(8)))  float  v8f;
typedef __attribute__((ext_vector_type(4)))  float  vf4;
typedef __attribute__((ext_vector_type(2)))  unsigned int v2u;
typedef __attribute__((ext_vector_type(4)))  unsigned int v4u;

union FragAB { v16bf v; v4u q[2]; };
union FragC  { v8f   v; float f[8]; };
union Pack2  { __bf16 h[2]; unsigned int u; };

// ---------------- generic strided GEMM: C[M,N] = act(A @ B [+ resid]) ----------------
// A element (m,k) at A[m*sAr + k*sAc]  (optional ReLU applied on load)
// B element (k,n) at B[k*sBr + n*sBc]
// C, resid row-major M x N.
// Block tile 256x128, BK=32; 8 waves; wave tile 64x64 => 16 WMMA / wave / K-step.
#define BM 256
#define BN 128
#define BK 32
#define LDSK 40   // 80B row stride: keeps ds b128 frag loads aligned & bank-conflict-free

__device__ __forceinline__ v2u cvt_quad(float v0, float v1, float v2, float v3, int reluA) {
  if (reluA) {
    v0 = fmaxf(v0, 0.0f); v1 = fmaxf(v1, 0.0f);
    v2 = fmaxf(v2, 0.0f); v3 = fmaxf(v3, 0.0f);
  }
  Pack2 p0, p1;
  p0.h[0] = (__bf16)v0; p0.h[1] = (__bf16)v1;
  p1.h[0] = (__bf16)v2; p1.h[1] = (__bf16)v3;
  v2u r; r[0] = p0.u; r[1] = p1.u;
  return r;
}

__global__ __launch_bounds__(256)
void gemm_bf16_wmma(const float* __restrict__ A, long long sAr, long long sAc, int reluA,
                    const float* __restrict__ B, long long sBr, long long sBc,
                    const float* __restrict__ resid,
                    float* __restrict__ C,
                    int M, int N, int K, int actRelu)
{
  __shared__ unsigned short As[BM][LDSK];   // [m][k] bf16
  __shared__ unsigned short Bs[BN][LDSK];   // [n][k] bf16 (B tile stored transposed)

  const int tid  = threadIdx.x;
  const int lane = tid & 31;
  const int wid  = tid >> 5;
  const int half = lane >> 4;     // K-half selector (wave32 WMMA layout)
  const int r    = lane & 15;

  const int m0 = blockIdx.y * BM;
  const int n0 = blockIdx.x * BN;

  const int waveRow = (wid >> 1) * 64;   // 4 wave-rows of 64
  const int waveCol = (wid & 1) * 64;    // 2 wave-cols of 64

  // staging coordinates: thread handles K-quads (4 f32 -> 2 packed uints -> one b64 DS store)
  const int rowQ = tid >> 3;          // 32 rows per e-step
  const int kq   = (tid & 7) * 4;     // k-quad base within tile

  const bool contigA = (sAc == 1);
  const bool contigB = (sBr == 1);

  FragC acc[4][4];
#pragma unroll
  for (int i = 0; i < 4; ++i)
#pragma unroll
    for (int j = 0; j < 4; ++j)
#pragma unroll
      for (int v = 0; v < 8; ++v) acc[i][j].f[v] = 0.0f;

  for (int k0 = 0; k0 < K; k0 += BK) {
    // ---- stage A tile (256x32): 8 quads/thread ----
    if (contigA) {
#pragma unroll
      for (int e = 0; e < 8; ++e) {
        const int row = rowQ + e * 32;
        const vf4 f = *reinterpret_cast<const vf4*>(&A[(long long)(m0 + row) * sAr + (k0 + kq)]);
        *reinterpret_cast<v2u*>(&As[row][kq]) = cvt_quad(f[0], f[1], f[2], f[3], reluA);
      }
    } else {
      const long long ka0 = (long long)(k0 + kq) * sAc;
#pragma unroll
      for (int e = 0; e < 8; ++e) {
        const int row = rowQ + e * 32;
        const long long base = (long long)(m0 + row) * sAr + ka0;
        *reinterpret_cast<v2u*>(&As[row][kq]) =
            cvt_quad(A[base], A[base + sAc], A[base + 2 * sAc], A[base + 3 * sAc], reluA);
      }
    }
    // ---- stage B tile transposed (128x32): 4 quads/thread ----
    if (contigB) {
#pragma unroll
      for (int e = 0; e < 4; ++e) {
        const int row = rowQ + e * 32;
        const vf4 f = *reinterpret_cast<const vf4*>(&B[(k0 + kq) + (long long)(n0 + row) * sBc]);
        *reinterpret_cast<v2u*>(&Bs[row][kq]) = cvt_quad(f[0], f[1], f[2], f[3], 0);
      }
    } else {
      const long long kb0 = (long long)(k0 + kq) * sBr;
#pragma unroll
      for (int e = 0; e < 4; ++e) {
        const int row = rowQ + e * 32;
        const long long base = kb0 + (long long)(n0 + row) * sBc;
        *reinterpret_cast<v2u*>(&Bs[row][kq]) =
            cvt_quad(B[base], B[base + sBr], B[base + 2 * sBr], B[base + 3 * sBr], 0);
      }
    }
    __syncthreads();

    // ---- prefetch next K-tile into cache (global_prefetch_b8, no VGPR cost) ----
    if (k0 + BK < K) {
      __builtin_prefetch(&A[(long long)(m0 + tid) * sAr + (long long)(k0 + BK) * sAc], 0, 3);
      __builtin_prefetch(&B[(long long)(k0 + BK) * sBr + (long long)(n0 + (tid & 127)) * sBc], 0, 3);
    }

    // ---- build fragments per ISA 16-bit A(16x32)/B(32x16) layouts ----
    // A frag: VGPRs 0-3 = K{0..7}+8*half, VGPRs 4-7 = K{16..23}+8*half  -> 2x ds_load_b128
    // B frag: VGPRs 0-3 = K{0..7}+8*half, VGPRs 4-7 = K{16..23}+8*half  -> 2x ds_load_b128
    FragAB fa[4], fb[4];
#pragma unroll
    for (int t = 0; t < 4; ++t) {
      const int row = waveRow + t * 16 + r;
      fa[t].q[0] = *reinterpret_cast<const v4u*>(&As[row][half * 8]);
      fa[t].q[1] = *reinterpret_cast<const v4u*>(&As[row][16 + half * 8]);
    }
#pragma unroll
    for (int t = 0; t < 4; ++t) {
      const int col = waveCol + t * 16 + r;
      fb[t].q[0] = *reinterpret_cast<const v4u*>(&Bs[col][half * 8]);
      fb[t].q[1] = *reinterpret_cast<const v4u*>(&Bs[col][16 + half * 8]);
    }

#pragma unroll
    for (int i = 0; i < 4; ++i)
#pragma unroll
      for (int j = 0; j < 4; ++j)
        acc[i][j].v = __builtin_amdgcn_wmma_f32_16x16x32_bf16(
            false, fa[i].v, false, fb[j].v, (short)0, acc[i][j].v, false, false);

    __syncthreads();
  }

  // ---- store: C/D layout — lane half h: VGPR v -> row v+8h, col r ----
  if (resid != nullptr) {
#pragma unroll
    for (int i = 0; i < 4; ++i) {
      const int mBase = m0 + waveRow + i * 16 + half * 8;
#pragma unroll
      for (int v = 0; v < 8; ++v) {
        const long long rowOff = (long long)(mBase + v) * N;
        float* __restrict__ cRow = C + rowOff;
        const float* __restrict__ rRow = resid + rowOff;
#pragma unroll
        for (int j = 0; j < 4; ++j) {
          const int n = n0 + waveCol + j * 16 + r;
          float val = acc[i][j].f[v] + rRow[n];
          if (actRelu) val = fmaxf(val, 0.0f);
          cRow[n] = val;
        }
      }
    }
  } else {
#pragma unroll
    for (int i = 0; i < 4; ++i) {
      const int mBase = m0 + waveRow + i * 16 + half * 8;
#pragma unroll
      for (int v = 0; v < 8; ++v) {
        float* __restrict__ cRow = C + (long long)(mBase + v) * N;
#pragma unroll
        for (int j = 0; j < 4; ++j) {
          const int n = n0 + waveCol + j * 16 + r;
          float val = acc[i][j].f[v];
          if (actRelu) val = fmaxf(val, 0.0f);
          cRow[n] = val;
        }
      }
    }
  }
}

// ---------------- fused residual + LayerNorm over last axis (SEQ) ----------------
__global__ __launch_bounds__(256)
void add_ln(const float* __restrict__ base, const float* __restrict__ delta,
            const float* __restrict__ w, const float* __restrict__ b,
            float* __restrict__ out)
{
  const int n = kSEQ;
  const int row = blockIdx.x;
  const int tid = threadIdx.x;
  __shared__ float red[256];

  float vals[4];
  float s = 0.0f;
#pragma unroll
  for (int e = 0; e < 4; ++e) {
    int idx = e * 256 + tid;
    float v = base[(long long)row * n + idx] + delta[(long long)row * n + idx];
    vals[e] = v; s += v;
  }
  red[tid] = s; __syncthreads();
  for (int off = 128; off > 0; off >>= 1) {
    if (tid < off) red[tid] += red[tid + off];
    __syncthreads();
  }
  const float mu = red[0] / (float)n;
  __syncthreads();

  float sq = 0.0f;
#pragma unroll
  for (int e = 0; e < 4; ++e) { float d = vals[e] - mu; sq += d * d; }
  red[tid] = sq; __syncthreads();
  for (int off = 128; off > 0; off >>= 1) {
    if (tid < off) red[tid] += red[tid + off];
    __syncthreads();
  }
  const float inv = rsqrtf(red[0] / (float)n + 1e-6f);

#pragma unroll
  for (int e = 0; e < 4; ++e) {
    int idx = e * 256 + tid;
    out[(long long)row * n + idx] = (vals[e] - mu) * inv * w[idx] + b[idx];
  }
}

// ---------------- host-side orchestration ----------------
static void launch_gemm(hipStream_t st,
                        const float* A, long long sAr, long long sAc, int reluA,
                        const float* B, long long sBr, long long sBc,
                        const float* resid, float* C,
                        int M, int N, int K, int act)
{
  dim3 grid((unsigned)(N / BN), (unsigned)(M / BM));
  gemm_bf16_wmma<<<grid, 256, 0, st>>>(A, sAr, sAc, reluA, B, sBr, sBc, resid, C, M, N, K, act);
}

struct Ws {
  float *rq, *hbuf, *sq, *bbuf, *att, *h2, *mhaout, *x1, *x2, *fmid;
};

static void run_mha(hipStream_t st, const float* x,
                    const float* wqkv, const float* W1, const float* W2,
                    const float* cW1, const float* cW2, const Ws& w)
{
  // rq[c] = relu(w_qkv[c] @ x)   (IN x SEQ)
  for (int c = 0; c < 3; ++c)
    launch_gemm(st, wqkv + (size_t)c * kIN * kIN, kIN, 1, 0,
                x, kSEQ, 1, nullptr,
                w.rq + (size_t)c * kIN * kSEQ, kIN, kSEQ, kIN, 1);

  for (int hd = 0; hd < kH; ++hd) {
    for (int c = 0; c < 3; ++c) {
      const float* W1p = W1 + (size_t)(hd * 3 + c) * kMID * kSEQ;  // (MID,SEQ)
      const float* W2p = W2 + (size_t)(hd * 3 + c) * kSEQ * kMID;  // (SEQ,MID)
      // h[i,m] = relu( sum_s rq[i,s] * W1[m,s] )
      launch_gemm(st, w.rq + (size_t)c * kIN * kSEQ, kSEQ, 1, 0,
                  W1p, 1, kSEQ, nullptr, w.hbuf, kIN, kMID, kSEQ, 1);
      // sq[i,s] = sum_m h[i,m] * W2[s,m]
      launch_gemm(st, w.hbuf, kMID, 1, 0,
                  W2p, 1, kMID, nullptr,
                  w.sq + (size_t)c * kIN * kSEQ, kIN, kSEQ, kMID, 0);
    }
    const float* q = w.sq;
    const float* k = w.sq + (size_t)kIN * kSEQ;
    const float* v = w.sq + (size_t)2 * kIN * kSEQ;
    // b[t,s] = sum_i k[i,t] * q[i,s]
    launch_gemm(st, k, 1, kSEQ, 0, q, kSEQ, 1, nullptr, w.bbuf, kSEQ, kSEQ, kIN, 0);
    // att[i,s] = sum_t v[i,t] * b[t,s]
    launch_gemm(st, v, kSEQ, 1, 0, w.bbuf, kSEQ, 1, nullptr,
                w.att + (size_t)hd * kIN * kSEQ, kIN, kSEQ, kSEQ, 0);
  }

  // h2[s,m] = relu( sum_j relu(cat[s,j]) * cW1[m,j] ), cat[s,j]=att[j,s]
  launch_gemm(st, w.att, 1, kSEQ, 1,
              cW1, 1, (long long)kH * kIN, nullptr,
              w.h2, kSEQ, kMID, kH * kIN, 1);
  // mha_out[i,s] = sum_m cW2[i,m] * h2[s,m]
  launch_gemm(st, cW2, kMID, 1, 0,
              w.h2, 1, kMID, nullptr, w.mhaout, kIN, kSEQ, kMID, 0);
}

extern "C" void kernel_launch(void* const* d_in, const int* in_sizes, int n_in,
                              void* d_out, int out_size, void* d_ws, size_t ws_size,
                              hipStream_t stream)
{
  (void)in_sizes; (void)n_in; (void)out_size; (void)ws_size;

  const float* inp   = (const float*)d_in[0];
  const float* wqkv1 = (const float*)d_in[1];
  const float* wqkv2 = (const float*)d_in[2];
  const float* m1W1  = (const float*)d_in[3];
  const float* m1W2  = (const float*)d_in[4];
  const float* m2W1  = (const float*)d_in[5];
  const float* m2W2  = (const float*)d_in[6];
  const float* c1W1  = (const float*)d_in[7];
  const float* c1W2  = (const float*)d_in[8];
  const float* c2W1  = (const float*)d_in[9];
  const float* c2W2  = (const float*)d_in[10];
  const float* l1W1  = (const float*)d_in[11];
  const float* l1W2  = (const float*)d_in[12];
  const float* l2W1  = (const float*)d_in[13];
  const float* l2W2  = (const float*)d_in[14];
  const float* nw    = (const float*)d_in[15];
  const float* nb    = (const float*)d_in[16];
  float* out = (float*)d_out;

  float* p = (float*)d_ws;
  Ws w;
  w.rq     = p; p += (size_t)3 * kIN * kSEQ;
  w.hbuf   = p; p += (size_t)kIN * kMID;
  w.sq     = p; p += (size_t)3 * kIN * kSEQ;
  w.bbuf   = p; p += (size_t)kSEQ * kSEQ;
  w.att    = p; p += (size_t)kH * kIN * kSEQ;
  w.h2     = p; p += (size_t)kSEQ * kMID;
  w.mhaout = p; p += (size_t)kIN * kSEQ;
  w.x1     = p; p += (size_t)kIN * kSEQ;
  w.x2     = p; p += (size_t)kIN * kSEQ;
  w.fmid   = p; p += (size_t)kSEQ * kMID;

  // layer 1: x1 = LN(inp + mha(inp))
  run_mha(stream, inp, wqkv1, m1W1, m1W2, c1W1, c1W2, w);
  add_ln<<<kIN, 256, 0, stream>>>(inp, w.mhaout, nw, nb, w.x1);

  // layer 2: x2 = LN(inp + mha(x1))   (residual is inp per reference)
  run_mha(stream, w.x1, wqkv2, m2W1, m2W2, c2W1, c2W2, w);
  add_ln<<<kIN, 256, 0, stream>>>(inp, w.mhaout, nw, nb, w.x2);

  // FFN heads: out = (relu(relu(x2^T) @ Wa^T) @ Wb^T)^T + x2
  launch_gemm(stream, w.x2, 1, kSEQ, 1, l1W1, 1, kIN, nullptr, w.fmid, kSEQ, kMID, kIN, 1);
  launch_gemm(stream, l1W2, kMID, 1, 0, w.fmid, 1, kMID, w.x2, out, kOUT, kSEQ, kMID, 0);

  launch_gemm(stream, w.x2, 1, kSEQ, 1, l2W1, 1, kIN, nullptr, w.fmid, kSEQ, kMID, kIN, 1);
  launch_gemm(stream, l2W2, kMID, 1, 0, w.fmid, 1, kMID, w.x2,
              out + (size_t)kOUT * kSEQ, kOUT, kSEQ, kMID, 0);
}
